// Attention_25572235280790
// MI455X (gfx1250) — compile-verified, matrix operand order checked by reading, not run on hardware
//
#include <hip/hip_runtime.h>
#include <hip/hip_bf16.h>

// ---------------------------------------------------------------------------
// Problem constants (match reference)
// ---------------------------------------------------------------------------
#define BB   16
#define CC   256
#define HH   32
#define WW   32
#define NHH  8
#define HD   32
#define LL   (HH * WW)          // 1024
#define MM   (BB * LL)          // 16384
#define SCALE_Q 0.17677669529663689f   // 32^-0.5
#define LN_EPS 1e-6f
#define BN_EPS 1e-5f

typedef __bf16 v16bf __attribute__((ext_vector_type(16)));
typedef float  v8f   __attribute__((ext_vector_type(8)));
typedef unsigned int v4u __attribute__((ext_vector_type(4)));
typedef int v4i __attribute__((ext_vector_type(4)));
typedef int v8i __attribute__((ext_vector_type(8)));

#if defined(__has_builtin)
# if __has_builtin(__builtin_amdgcn_tensor_load_to_lds)
#  define HAVE_TDM 1
# endif
#endif

__device__ __forceinline__ unsigned short f2bf(float f) {
  unsigned u = __float_as_uint(f);
  u += 0x7FFFu + ((u >> 16) & 1u);
  return (unsigned short)(u >> 16);
}
__device__ __forceinline__ float bf2f(unsigned short h) {
  return __uint_as_float(((unsigned)h) << 16);
}
__device__ __forceinline__ v8f vzero8() {
  v8f z;
#pragma unroll
  for (int i = 0; i < 8; ++i) z[i] = 0.0f;
  return z;
}
// A/B fragment: lane holds row (or col) r = lane%16, halves at K offsets
// {base..base+7, base+16..base+23}, base = (lane<16)?0:8  (ISA 7.12.2)
__device__ __forceinline__ v16bf load_frag(const unsigned short* row, int kbase) {
  union { v16bf v; uint4 q[2]; } u;
  u.q[0] = *(const uint4*)(row + kbase);
  u.q[1] = *(const uint4*)(row + kbase + 16);
  return u.v;
}
__device__ __forceinline__ v8f wmma_bf16(v16bf a, v16bf b, v8f c) {
  return __builtin_amdgcn_wmma_f32_16x16x32_bf16(false, a, false, b,
                                                 (short)0, c, false, false);
}
// ASYNCcnt path: global -> LDS, 16B per lane, no VGPR round-trip.
__device__ __forceinline__ void async_b128(const void* g, void* lds) {
  asm volatile("global_load_async_to_lds_b128 %0, %1, off"
               :: "v"((unsigned)(uintptr_t)lds), "v"(g)
               : "memory");
}
__device__ __forceinline__ void wait_async0() {
  asm volatile("s_wait_asynccnt 0" ::: "memory");
}

// ---------------------------------------------------------------------------
// Prep: x (B,C,H,W) fp32 -> Xbf [M=B*H*W, C] bf16 (position-major)
// ---------------------------------------------------------------------------
__global__ __launch_bounds__(256) void transpose_x_kernel(
    const float* __restrict__ x, unsigned short* __restrict__ Xbf) {
  int idx = blockIdx.x * 256 + threadIdx.x;   // m*C + c
  int c = idx & (CC - 1);
  int m = idx >> 8;
  int b  = m >> 10;
  int hw = m & (LL - 1);
  Xbf[idx] = f2bf(x[(b * CC + c) * LL + hw]);
}

__global__ __launch_bounds__(256) void cvt_bf16_kernel(
    const float* __restrict__ in, unsigned short* __restrict__ out, int n) {
  int i = blockIdx.x * 256 + threadIdx.x;
  if (i < n) out[i] = f2bf(in[i]);
}

// ---------------------------------------------------------------------------
// WMMA GEMM: Y[M,N] = Xbf[M,256] * W[N,256]^T (+ epilogue)
// Block 256 threads (8 waves), tile BM=128 x BN=64, BK=32, waves 4x2,
// each wave computes 32x32 via 2x2 wmma_f32_16x16x32_bf16.
// Double-buffered: async-to-LDS DMA for tile i+1 overlaps WMMA on tile i;
// one barrier per k-step.
// MODE 0: +bias, ->bf16 store (qkv).  MODE 1: +bias, BN affine, SiLU -> f32.
// MODE 2: +bias, scatter to NCHW f32 (final output).
// ---------------------------------------------------------------------------
__device__ __forceinline__ void gemm_stage(const unsigned short* A,
                                           const unsigned short* Wb, int m0,
                                           int n0, int k0, int tid,
                                           unsigned short* At,
                                           unsigned short* Bt) {
  int rowA = tid >> 1, offA = (tid & 1) * 16;
  const unsigned short* g = A + (size_t)(m0 + rowA) * 256 + k0 + offA;
  async_b128(g, At + rowA * 40 + offA);
  async_b128(g + 8, At + rowA * 40 + offA + 8);
  int rowB = tid >> 2, offB = (tid & 3) * 8;
  async_b128(Wb + (size_t)(n0 + rowB) * 256 + k0 + offB, Bt + rowB * 40 + offB);
}

template <int MODE>
__global__ __launch_bounds__(256) void gemm_bf16_kernel(
    const unsigned short* __restrict__ A,   // [M,256] bf16
    const unsigned short* __restrict__ Wb,  // [N,256] bf16
    const float* __restrict__ bias,         // [N]
    const float* __restrict__ e0,           // bn_gamma (MODE 1)
    const float* __restrict__ e1,           // bn_beta  (MODE 1)
    void* __restrict__ out, int N) {
  __shared__ __align__(16) unsigned short At[2][128 * 40];
  __shared__ __align__(16) unsigned short Bt[2][64 * 40];

  const int m0 = blockIdx.x * 128;
  const int n0 = blockIdx.y * 64;
  const int tid = threadIdx.x;
  const int w = tid >> 5, lane = tid & 31;
  const int wm = w & 3, wn = w >> 2;
  const int hl = lane >> 4, ln = lane & 15;
  const int kbase = hl ? 8 : 0;

  v8f acc[4];
#pragma unroll
  for (int t = 0; t < 4; ++t) acc[t] = vzero8();

  gemm_stage(A, Wb, m0, n0, 0, tid, At[0], Bt[0]);  // prologue

  for (int it = 0; it < 8; ++it) {
    const int cur = it & 1;
    wait_async0();       // my DMA into buf[cur] done
    __syncthreads();     // everyone's done; buf[cur^1] readers finished
    if (it < 7)          // prefetch next tile while computing this one
      gemm_stage(A, Wb, m0, n0, (it + 1) * 32, tid, At[cur ^ 1], Bt[cur ^ 1]);

    v16bf a0 = load_frag(At[cur] + (wm * 32 + ln) * 40, kbase);
    v16bf a1 = load_frag(At[cur] + (wm * 32 + 16 + ln) * 40, kbase);
    v16bf b0 = load_frag(Bt[cur] + (wn * 32 + ln) * 40, kbase);
    v16bf b1 = load_frag(Bt[cur] + (wn * 32 + 16 + ln) * 40, kbase);

    acc[0] = wmma_bf16(a0, b0, acc[0]);
    acc[1] = wmma_bf16(a0, b1, acc[1]);
    acc[2] = wmma_bf16(a1, b0, acc[2]);
    acc[3] = wmma_bf16(a1, b1, acc[3]);
  }

  const float bn_rsq = rsqrtf(1.0f + BN_EPS);
#pragma unroll
  for (int t = 0; t < 4; ++t) {
    int sa = t >> 1, sb = t & 1;
    int gn = n0 + wn * 32 + sb * 16 + ln;
    float bs = bias[gn];
#pragma unroll
    for (int i = 0; i < 8; ++i) {
      int gm = m0 + wm * 32 + sa * 16 + i + 8 * hl;
      float v = acc[t][i] + bs;
      if (MODE == 0) {
        ((unsigned short*)out)[(size_t)gm * N + gn] = f2bf(v);
      } else if (MODE == 1) {
        float g = v * (e0[gn] * bn_rsq) + e1[gn];
        float s = g / (1.0f + __expf(-g));  // SiLU
        ((float*)out)[(size_t)gm * 256 + gn] = s;
      } else {  // MODE 2: scatter to (B,C,H,W)
        int b = gm >> 10, hw = gm & (LL - 1);
        ((float*)out)[((size_t)b * CC + gn) * LL + hw] = v;
      }
    }
  }
}

// ---------------------------------------------------------------------------
// RoPE on q,k (+scale on q), repack q/k/v -> [B*NH, L, HD] bf16
// ---------------------------------------------------------------------------
__global__ __launch_bounds__(256) void rope_kernel(
    const unsigned short* __restrict__ qkv,  // [M,768] bf16
    const float* __restrict__ sinp, const float* __restrict__ cosp,
    unsigned short* __restrict__ Qo, unsigned short* __restrict__ Ko,
    unsigned short* __restrict__ Vo) {
  int idx = blockIdx.x * 256 + threadIdx.x;  // B*NH*L*16 (one d-pair each)
  int j  = idx & 15;
  int l  = (idx >> 4) & (LL - 1);
  int nh = (idx >> 14) & 7;
  int b  = idx >> 17;
  int d  = j * 2;
  size_t m = (size_t)b * LL + l;
  int cq = nh * HD + d;

  float q0 = bf2f(qkv[m * 768 + cq]);
  float q1 = bf2f(qkv[m * 768 + cq + 1]);
  float k0 = bf2f(qkv[m * 768 + 256 + cq]);
  float k1 = bf2f(qkv[m * 768 + 256 + cq + 1]);
  float s0 = sinp[l * HD + d], s1 = sinp[l * HD + d + 1];
  float c0 = cosp[l * HD + d], c1 = cosp[l * HD + d + 1];

  float qe = (q0 * c0 - q1 * s0) * SCALE_Q;
  float qo = (q1 * c1 + q0 * s1) * SCALE_Q;
  float ke = k0 * c0 - k1 * s0;
  float ko = k1 * c1 + k0 * s1;

  size_t o = ((size_t)(b * NHH + nh) * LL + l) * HD + d;
  Qo[o] = f2bf(qe); Qo[o + 1] = f2bf(qo);
  Ko[o] = f2bf(ke); Ko[o + 1] = f2bf(ko);
  Vo[o]     = qkv[m * 768 + 512 + cq];
  Vo[o + 1] = qkv[m * 768 + 512 + cq + 1];
}

// ---------------------------------------------------------------------------
// Depthwise 5x5 conv (lepe) on v, zero-padded; -> [M, C] f32 (position-major)
// ---------------------------------------------------------------------------
__global__ __launch_bounds__(256) void lepe_kernel(
    const unsigned short* __restrict__ qkv, const float* __restrict__ wl,
    const float* __restrict__ bl, float* __restrict__ lepe) {
  int idx = blockIdx.x * 256 + threadIdx.x;  // m*C + c
  int c = idx & (CC - 1);
  int m = idx >> 8;
  int hw = m & (LL - 1);
  int b = m >> 10;
  int h = hw >> 5, wx = hw & 31;
  float acc = bl[c];
#pragma unroll
  for (int dy = -2; dy <= 2; ++dy) {
    int hh = h + dy;
    if ((unsigned)hh >= (unsigned)HH) continue;
#pragma unroll
    for (int dx = -2; dx <= 2; ++dx) {
      int ww2 = wx + dx;
      if ((unsigned)ww2 >= (unsigned)WW) continue;
      float vv = bf2f(qkv[((size_t)b * LL + hh * WW + ww2) * 768 + 512 + c]);
      acc += vv * wl[c * 25 + (dy + 2) * 5 + (dx + 2)];
    }
  }
  lepe[idx] = acc;
}

// ---------------------------------------------------------------------------
// Flash attention per (b, nh): L=1024, HD=32 (HD == WMMA K exactly).
// Transposed-score formulation: ST = K_tile x Q^T so that softmax row
// reductions are IN-LANE and the P^T C-layout equals the A-fragment layout
// for P@V (no LDS round-trip for P, no per-tile bpermute trees).
// Double-buffered K/V staging: K tile via Tensor Data Mover (pad 64B+16B ->
// conflict-free 80B LDS rows) or async-to-LDS fallback; V staged transposed.
// One barrier per k-tile; DMA for tile i+1 overlaps WMMA on tile i.
// ---------------------------------------------------------------------------
__global__ __launch_bounds__(128) void attn_kernel(
    const unsigned short* __restrict__ Q, const unsigned short* __restrict__ K,
    const unsigned short* __restrict__ V, float* __restrict__ O) {
  __shared__ __align__(16) unsigned short Kt[2][32 * 40];
  __shared__ __align__(16) unsigned short Vt[2][32 * 40];

  const int bh = blockIdx.x;            // b*NH + nh
  const int b = bh >> 3, nh = bh & 7;
  const int q0 = blockIdx.y * 64;
  const int tid = threadIdx.x;
  const int w = tid >> 5, lane = tid & 31;
  const int hl = lane >> 4, ln = lane & 15;
  const int kbase = hl ? 8 : 0;

  // Q as B-fragment (col q = ln, elements d), loaded once from global
  const unsigned short* Qrow = Q + ((size_t)bh * LL + q0 + w * 16 + ln) * HD;
  union { v16bf v; uint4 q[2]; } qb;
  qb.q[0] = *(const uint4*)(Qrow + kbase);
  qb.q[1] = *(const uint4*)(Qrow + kbase + 16);

  float rmax = -3.0e38f, rsum = 0.0f;   // per-lane: state of q-row `ln`
  v8f oa = vzero8(), ob = vzero8();

  // --- staging helpers -----------------------------------------------------
  auto stage_v = [&](int buf, int kt) {  // V transposed [d][kpos]
    int row = tid >> 2, offd = (tid & 3) * 8;
    uint4 dv = *(const uint4*)(V + ((size_t)bh * LL + kt + row) * HD + offd);
    const unsigned short* pv = (const unsigned short*)&dv;
#pragma unroll
    for (int jj = 0; jj < 8; ++jj) Vt[buf][(offd + jj) * 40 + row] = pv[jj];
  };
  auto stage_k = [&](int buf, int kt) {
#ifdef HAVE_TDM
    if (w == 0) {  // one TDM descriptor moves the whole padded K tile
      unsigned long long ga =
          (unsigned long long)(uintptr_t)(K + ((size_t)bh * LL + kt) * HD);
      unsigned ldsa = (unsigned)(uintptr_t)Kt[buf];
      v4u g0 = { 1u,                               // count=1, load, no gather
                 ldsa,                             // lds_addr
                 (unsigned)ga,                     // global_addr[31:0]
                 (unsigned)((ga >> 32) & 0x01FFFFFFu) | (2u << 30) };  // +type
      // data_size=2B, pad_enable, pad every 16 DW (64B) by 4 DW (16B)
      v8i g1 = { (int)0x06D10000,
                 (int)(32u << 16),                 // tensor_dim0 = 32
                 (int)(32u << 16),                 // tensor_dim1 = 32
                 (int)(32u << 16),                 // tile_dim0 = 32
                 32,                               // tile_dim1 = 32
                 32,                               // tensor_dim0_stride = 32
                 (int)(1024u << 16),               // tensor_dim1_stride lo
                 0 };
      v4i gz = { 0, 0, 0, 0 };
#if __clang_major__ >= 23
      v8i gz8 = { 0, 0, 0, 0, 0, 0, 0, 0 };
      __builtin_amdgcn_tensor_load_to_lds(g0, g1, gz, gz, gz8, 0);
#else
      __builtin_amdgcn_tensor_load_to_lds(g0, g1, gz, gz, 0);
#endif
    }
#else
    int row = tid >> 2, offd = (tid & 3) * 8;
    async_b128(K + ((size_t)bh * LL + kt + row) * HD + offd,
               Kt[buf] + row * 40 + offd);
#endif
  };
  auto wait_k = [&]() {
#ifdef HAVE_TDM
    if (w == 0) asm volatile("s_wait_tensorcnt 0" ::: "memory");
#else
    wait_async0();
#endif
  };
  // -------------------------------------------------------------------------

  stage_v(0, 0);
  stage_k(0, 0);

  for (int it = 0; it < LL / 32; ++it) {
    const int cur = it & 1;
    const int kt = it * 32;
    wait_k();            // DMA into Kt[cur] complete (issuing wave)
    __syncthreads();     // tiles[cur] ready; tiles[cur^1] readers finished
    if (it < LL / 32 - 1) {  // prefetch next tile during compute
      stage_v(cur ^ 1, kt + 32);
      stage_k(cur ^ 1, kt + 32);
    }

    // ST = K_tile x Q^T : two 16(kpos) x 16(q) tiles
    v16bf ak0 = load_frag(Kt[cur] + ln * 40, kbase);
    v16bf ak1 = load_frag(Kt[cur] + (16 + ln) * 40, kbase);
    v8f st0 = wmma_bf16(ak0, qb.v, vzero8());
    v8f st1 = wmma_bf16(ak1, qb.v, vzero8());

    // in-lane softmax over this lane's 16 kpos values + one xor16 pairing
    float tmax = st0[0];
#pragma unroll
    for (int i = 1; i < 8; ++i) tmax = fmaxf(tmax, st0[i]);
#pragma unroll
    for (int i = 0; i < 8; ++i) tmax = fmaxf(tmax, st1[i]);
    tmax = fmaxf(tmax, __shfl_xor(tmax, 16, 32));
    float nm = fmaxf(rmax, tmax);
    float corr = __expf(rmax - nm);
    rmax = nm;

    float p0[8], p1[8], ls = 0.0f;
#pragma unroll
    for (int i = 0; i < 8; ++i) {
      p0[i] = __expf(st0[i] - nm);
      p1[i] = __expf(st1[i] - nm);
      ls += p0[i] + p1[i];
    }
    ls += __shfl_xor(ls, 16, 32);
    rsum = rsum * corr + ls;

    // P^T C-layout == A-frag layout: pack in registers, no LDS round-trip
    union { v16bf v; __bf16 e[16]; } pf;
#pragma unroll
    for (int i = 0; i < 8; ++i) {
      pf.e[i]     = (__bf16)p0[i];
      pf.e[8 + i] = (__bf16)p1[i];
    }

    // broadcast corr to the O rows this lane accumulates (q = i + 8*hl)
#pragma unroll
    for (int i = 0; i < 8; ++i) {
      float cb = __shfl(corr, i + 8 * hl, 32);
      oa[i] *= cb;
      ob[i] *= cb;
    }

    v16bf bv0 = load_frag(Vt[cur] + ln * 40, kbase);
    v16bf bv1 = load_frag(Vt[cur] + (16 + ln) * 40, kbase);
    oa = wmma_bf16(pf.v, bv0, oa);
    ob = wmma_bf16(pf.v, bv1, ob);
  }

#pragma unroll
  for (int i = 0; i < 8; ++i) {
    float rs = __shfl(rsum, i + 8 * hl, 32);
    float inv = 1.0f / rs;
    int l = q0 + w * 16 + i + 8 * hl;
    size_t m = (size_t)b * LL + l;
    O[m * CC + nh * HD + ln] = oa[i] * inv;
    O[m * CC + nh * HD + 16 + ln] = ob[i] * inv;
  }
}

// ---------------------------------------------------------------------------
// Fused: (O + lepe) -> LN1(norm) * gate -> LN2(ln) -> bf16 [M, C]
// One wave per position; lane holds 8 contiguous channels.
// ---------------------------------------------------------------------------
__global__ __launch_bounds__(256) void ln_fuse_kernel(
    const float* __restrict__ O, const float* __restrict__ lepe,
    const float* __restrict__ gate, const float* __restrict__ ng,
    const float* __restrict__ nb, const float* __restrict__ lg,
    const float* __restrict__ lb, unsigned short* __restrict__ Xo) {
  int w = threadIdx.x >> 5, lane = threadIdx.x & 31;
  size_t pos = (size_t)blockIdx.x * 8 + w;
  int cb = lane * 8;
  const float* po = O + pos * CC + cb;
  const float* pl = lepe + pos * CC + cb;
  const float* pg = gate + pos * CC + cb;

  float v[8], s = 0.0f;
#pragma unroll
  for (int j = 0; j < 8; ++j) { v[j] = po[j] + pl[j]; s += v[j]; }
  s += __shfl_xor(s, 1, 32); s += __shfl_xor(s, 2, 32);
  s += __shfl_xor(s, 4, 32); s += __shfl_xor(s, 8, 32);
  s += __shfl_xor(s, 16, 32);
  float mu = s * (1.0f / CC);
  float vs = 0.0f;
#pragma unroll
  for (int j = 0; j < 8; ++j) { float d = v[j] - mu; vs += d * d; }
  vs += __shfl_xor(vs, 1, 32); vs += __shfl_xor(vs, 2, 32);
  vs += __shfl_xor(vs, 4, 32); vs += __shfl_xor(vs, 8, 32);
  vs += __shfl_xor(vs, 16, 32);
  float rstd = rsqrtf(vs * (1.0f / CC) + LN_EPS);

  float t[8], s2 = 0.0f;
#pragma unroll
  for (int j = 0; j < 8; ++j) {
    t[j] = ((v[j] - mu) * rstd * ng[cb + j] + nb[cb + j]) * pg[j];
    s2 += t[j];
  }
  s2 += __shfl_xor(s2, 1, 32); s2 += __shfl_xor(s2, 2, 32);
  s2 += __shfl_xor(s2, 4, 32); s2 += __shfl_xor(s2, 8, 32);
  s2 += __shfl_xor(s2, 16, 32);
  float mu2 = s2 * (1.0f / CC);
  float vs2 = 0.0f;
#pragma unroll
  for (int j = 0; j < 8; ++j) { float d = t[j] - mu2; vs2 += d * d; }
  vs2 += __shfl_xor(vs2, 1, 32); vs2 += __shfl_xor(vs2, 2, 32);
  vs2 += __shfl_xor(vs2, 4, 32); vs2 += __shfl_xor(vs2, 8, 32);
  vs2 += __shfl_xor(vs2, 16, 32);
  float rstd2 = rsqrtf(vs2 * (1.0f / CC) + LN_EPS);
#pragma unroll
  for (int j = 0; j < 8; ++j)
    Xo[pos * CC + cb + j] =
        f2bf((t[j] - mu2) * rstd2 * lg[cb + j] + lb[cb + j]);
}

// ---------------------------------------------------------------------------
// Launch
// ---------------------------------------------------------------------------
extern "C" void kernel_launch(void* const* d_in, const int* in_sizes, int n_in,
                              void* d_out, int out_size, void* d_ws,
                              size_t ws_size, hipStream_t stream) {
  const float* x        = (const float*)d_in[0];
  const float* sinp     = (const float*)d_in[1];
  const float* cosp     = (const float*)d_in[2];
  const float* w_qkv    = (const float*)d_in[3];
  const float* b_qkv    = (const float*)d_in[4];
  const float* w_gate   = (const float*)d_in[5];
  const float* b_gate   = (const float*)d_in[6];
  const float* bn_gamma = (const float*)d_in[7];
  const float* bn_beta  = (const float*)d_in[8];
  const float* w_lepe   = (const float*)d_in[9];
  const float* b_lepe   = (const float*)d_in[10];
  const float* norm_g   = (const float*)d_in[11];
  const float* norm_b   = (const float*)d_in[12];
  const float* ln_g     = (const float*)d_in[13];
  const float* ln_b     = (const float*)d_in[14];
  const float* w_proj   = (const float*)d_in[15];
  const float* b_proj   = (const float*)d_in[16];

  char* ws = (char*)d_ws;
  size_t off = 0;
  auto alloc = [&](size_t bytes) {
    void* p = ws + off;
    off += (bytes + 255) & ~(size_t)255;
    return p;
  };
  unsigned short* Xbf   = (unsigned short*)alloc((size_t)MM * CC * 2);
  unsigned short* Wqkvb = (unsigned short*)alloc((size_t)768 * 256 * 2);
  unsigned short* Wgb   = (unsigned short*)alloc((size_t)256 * 256 * 2);
  unsigned short* Wpb   = (unsigned short*)alloc((size_t)256 * 256 * 2);
  unsigned short* qkvb  = (unsigned short*)alloc((size_t)MM * 768 * 2);
  float*          gateb = (float*)alloc((size_t)MM * CC * 4);
  unsigned short* Qb    = (unsigned short*)alloc((size_t)MM * CC * 2);
  unsigned short* Kb    = (unsigned short*)alloc((size_t)MM * CC * 2);
  unsigned short* Vb    = (unsigned short*)alloc((size_t)MM * CC * 2);
  float*          Ob    = (float*)alloc((size_t)MM * CC * 4);
  float*          lepeb = (float*)alloc((size_t)MM * CC * 4);
  unsigned short* Xbf2  = (unsigned short*)alloc((size_t)MM * CC * 2);

  // 1) layout transform + weight conversion
  transpose_x_kernel<<<(MM * CC) / 256, 256, 0, stream>>>(x, Xbf);
  cvt_bf16_kernel<<<(768 * 256 + 255) / 256, 256, 0, stream>>>(w_qkv, Wqkvb, 768 * 256);
  cvt_bf16_kernel<<<(256 * 256 + 255) / 256, 256, 0, stream>>>(w_gate, Wgb, 256 * 256);
  cvt_bf16_kernel<<<(256 * 256 + 255) / 256, 256, 0, stream>>>(w_proj, Wpb, 256 * 256);

  // 2) gate = SiLU(BN(X * Wg^T + b))   [MODE 1]
  gemm_bf16_kernel<1><<<dim3(MM / 128, 256 / 64), 256, 0, stream>>>(
      Xbf, Wgb, b_gate, bn_gamma, bn_beta, (void*)gateb, 256);

  // 3) qkv = X * Wqkv^T + b  -> bf16   [MODE 0]
  gemm_bf16_kernel<0><<<dim3(MM / 128, 768 / 64), 256, 0, stream>>>(
      Xbf, Wqkvb, b_qkv, nullptr, nullptr, (void*)qkvb, 768);

  // 4) RoPE + repack q/k/v
  rope_kernel<<<(BB * NHH * LL * 16) / 256, 256, 0, stream>>>(
      qkvb, sinp, cosp, Qb, Kb, Vb);

  // 5) lepe depthwise conv
  lepe_kernel<<<(MM * CC) / 256, 256, 0, stream>>>(qkvb, w_lepe, b_lepe, lepeb);

  // 6) flash attention (transposed-score WMMA, double-buffered staging)
  attn_kernel<<<dim3(BB * NHH, LL / 64), 128, 0, stream>>>(Qb, Kb, Vb, Ob);

  // 7) fused (O + lepe) -> LN1 * gate -> LN2 -> bf16
  ln_fuse_kernel<<<MM / 8, 256, 0, stream>>>(Ob, lepeb, gateb, norm_g, norm_b,
                                             ln_g, ln_b, Xbf2);

  // 8) proj GEMM -> NCHW output       [MODE 2]
  gemm_bf16_kernel<2><<<dim3(MM / 128, 256 / 64), 256, 0, stream>>>(
      Xbf2, Wpb, b_proj, nullptr, nullptr, d_out, 256);
}
